// Attention_63015760167295
// MI455X (gfx1250) — compile-verified, matrix operand order checked by reading, not run on hardware
//
#include <hip/hip_runtime.h>
#include <hip/hip_bf16.h>

typedef __bf16 bf16;
typedef __attribute__((ext_vector_type(16))) __bf16 v16bf;
typedef __attribute__((ext_vector_type(8)))  __bf16 v8bf;
typedef __attribute__((ext_vector_type(8)))  float  v8f;

#define WMMA_BF16(a, b, c) \
    __builtin_amdgcn_wmma_f32_16x16x32_bf16(false, (a), false, (b), (short)0, (c), false, false)

// ---------------------------------------------------------------------------
// Fragment loader: 16x32 bf16 tile, row-major with stride `ld` (elements).
// A-fragment: lane = row (lane&15), element chunks = K [h*8, h*8+8) and
// [16+h*8, 16+h*8+8) where h = lane>>4  (ISA 7.12.2, 16-bit A 16x32).
// Identical pattern serves as the B-fragment from a row-major [N, K] matrix.
// ---------------------------------------------------------------------------
__device__ __forceinline__ v16bf load_frag_rm(const bf16* base, int ld) {
    const int lane = threadIdx.x & 31;
    const int r = lane & 15;
    const int h = lane >> 4;
    const bf16* p = base + r * ld + h * 8;
    union { v16bf v; v8bf half8[2]; } u;
    u.half8[0] = *(const v8bf*)(p);        // K = h*8 .. h*8+7
    u.half8[1] = *(const v8bf*)(p + 16);   // K = 16+h*8 .. 16+h*8+7
    return u.v;
}

// ---------------------------------------------------------------------------
// CDNA5 async global->LDS copy (tracked by ASYNCcnt), via inline asm so it
// works on both toolchains regardless of builtin arity.
// ---------------------------------------------------------------------------
__device__ __forceinline__ uint32_t lds_off_of(const void* p) {
    return (uint32_t)(uintptr_t)(__attribute__((address_space(3))) const void*)p;
}

__device__ __forceinline__ void async_copy_b128(uint32_t lds_byte_off,
                                                const void* gptr) {
    asm volatile("global_load_async_to_lds_b128 %0, %1, off"
                 :: "v"(lds_byte_off), "v"((uint64_t)(uintptr_t)gptr)
                 : "memory");
}

__device__ __forceinline__ void wait_async0() {
    asm volatile("s_wait_asynccnt 0" ::: "memory");
}

// ---------------------------------------------------------------------------
// f32 -> bf16 conversion
// ---------------------------------------------------------------------------
__global__ void cvt_bf16_kernel(const float* __restrict__ src,
                                bf16* __restrict__ dst, int n) {
    int i = blockIdx.x * blockDim.x + threadIdx.x;
    if (i < n) dst[i] = (bf16)src[i];
}

// ---------------------------------------------------------------------------
// QKV GEMM: [16384, 768] x [2304, 768]^T -> scatter into Q/K/Vt (bf16).
// Block = 128 threads = 4 waves; each wave computes a 32(M) x 64(N) tile.
// Q,K layout: [B*H, N, 64] row-major. Vt layout: [B*H, 64, N] row-major.
// ---------------------------------------------------------------------------
__global__ __launch_bounds__(128) void qkv_gemm_kernel(
        const bf16* __restrict__ xb, const bf16* __restrict__ wb,
        bf16* __restrict__ Qb, bf16* __restrict__ Kb, bf16* __restrict__ Vtb) {
    const int wave = threadIdx.x >> 5;
    const int lane = threadIdx.x & 31;
    const int m0 = blockIdx.x * 128 + wave * 32; // token rows
    const int n0 = blockIdx.y * 64;              // output features

    v8f acc[2][4] = {{v8f{}, v8f{}, v8f{}, v8f{}},
                     {v8f{}, v8f{}, v8f{}, v8f{}}};
    for (int k = 0; k < 768; k += 32) {
        v16bf a0 = load_frag_rm(xb + (size_t)m0 * 768 + k, 768);
        v16bf a1 = load_frag_rm(xb + (size_t)(m0 + 16) * 768 + k, 768);
#pragma unroll
        for (int nt = 0; nt < 4; ++nt) {
            v16bf b = load_frag_rm(wb + (size_t)(n0 + nt * 16) * 768 + k, 768);
            acc[0][nt] = WMMA_BF16(a0, b, acc[0][nt]);
            acc[1][nt] = WMMA_BF16(a1, b, acc[1][nt]);
        }
    }

    const int nl = lane & 15;
    const int hl = lane >> 4;
#pragma unroll
    for (int mt = 0; mt < 2; ++mt) {
#pragma unroll
        for (int nt = 0; nt < 4; ++nt) {
            const int c = n0 + nt * 16 + nl;     // column in [0, 2304)
            const int three = c / 768;           // 0=Q, 1=K, 2=V
            const int rem = c - three * 768;
            const int head = rem >> 6;
            const int d = rem & 63;
#pragma unroll
            for (int v = 0; v < 8; ++v) {
                const int t = m0 + mt * 16 + v + 8 * hl;  // global token
                const int b = t >> 10;
                const int nq = t & 1023;
                const size_t bh = (size_t)(b * 12 + head);
                const bf16 val = (bf16)acc[mt][nt][v];
                if (three == 0)      Qb[(bh * 1024 + nq) * 64 + d] = val;
                else if (three == 1) Kb[(bh * 1024 + nq) * 64 + d] = val;
                else                 Vtb[(bh * 64 + d) * 1024 + nq] = val;
            }
        }
    }
}

// ---------------------------------------------------------------------------
// Flash attention: grid (B*H, N/64), 128 threads (4 waves), 16 queries/wave.
// The 64-key K and V^T blocks are shared by all 4 waves, so they are staged
// into LDS with double-buffered async global->LDS copies (ASYNCcnt) issued
// one block ahead; fragments then come from ds_load_b128.
// ---------------------------------------------------------------------------
__global__ __launch_bounds__(128) void attn_kernel(
        const bf16* __restrict__ Qb, const bf16* __restrict__ Kb,
        const bf16* __restrict__ Vtb, bf16* __restrict__ attnb) {
    const int bh = blockIdx.x;            // 0..191
    const int tid = threadIdx.x;
    const int wave = tid >> 5;
    const int lane = tid & 31;
    const int q0 = blockIdx.y * 64 + wave * 16;   // query row within N

    const bf16* Q  = Qb  + (size_t)bh * 1024 * 64;
    const bf16* K  = Kb  + (size_t)bh * 1024 * 64;
    const bf16* Vt = Vtb + (size_t)bh * 64 * 1024;

    __shared__ __align__(16) bf16 Kt[2][64 * 64];    // 64 keys x 64 dims
    __shared__ __align__(16) bf16 Vtt[2][64 * 64];   // 64 dims  x 64 keys
    __shared__ __align__(16) bf16 plds[4][16 * 64];  // per-wave P staging
    bf16* pw = plds[wave];

    // Stage one 64-key block: 512 x 16B chunks for K + same for V^T,
    // 4 chunks of each per thread.
    auto stage = [&](int buf, int kb) {
#pragma unroll
        for (int swp = 0; swp < 4; ++swp) {
            const int c = tid + swp * 128;     // 0..511
            const int row = c >> 2;            // 0..127
            const int col = (c & 3) * 16;      // element offset, 16 bf16 chunks
            async_copy_b128(lds_off_of(&Kt[buf][row * 64 + col]),
                            K + (size_t)(kb + row) * 64 + col);
            async_copy_b128(lds_off_of(&Vtt[buf][row * 64 + col]),
                            Vt + (size_t)row * 1024 + kb + col);
        }
    };

    // Q fragments resident in registers (hd = 64 -> two K=32 chunks)
    const v16bf qa0 = load_frag_rm(Q + (size_t)q0 * 64, 64);
    const v16bf qa1 = load_frag_rm(Q + (size_t)q0 * 64 + 32, 64);

    v8f o[4] = {v8f{}, v8f{}, v8f{}, v8f{}};
    float rm[8], rs[8];
#pragma unroll
    for (int v = 0; v < 8; ++v) { rm[v] = -1e30f; rs[v] = 0.0f; }

    const float scale = 0.125f;           // 1/sqrt(64)
    const int nl = lane & 15;
    const int hl = lane >> 4;

    stage(0, 0);

    for (int kb = 0; kb < 1024; kb += 64) {
        const int cur = (kb >> 6) & 1;

        wait_async0();          // my copies into Kt/Vtt[cur] are done
        __syncthreads();        // everyone's copies are done

        if (kb + 64 < 1024) stage(cur ^ 1, kb + 64);

        const bf16* Kc = Kt[cur];
        const bf16* Vc = Vtt[cur];

        // S = Q * K^T for 64 keys: four 16x16 accumulators
        v8f s[4] = {v8f{}, v8f{}, v8f{}, v8f{}};
#pragma unroll
        for (int i = 0; i < 4; ++i) {
            v16bf kf = load_frag_rm(Kc + (size_t)(i * 16) * 64, 64);
            s[i] = WMMA_BF16(qa0, kf, s[i]);
            kf = load_frag_rm(Kc + (size_t)(i * 16) * 64 + 32, 64);
            s[i] = WMMA_BF16(qa1, kf, s[i]);
        }

        // Online softmax per row (row m = v + 8*hl)
        float p[4][8];
#pragma unroll
        for (int v = 0; v < 8; ++v) {
            float sv0 = s[0][v] * scale;
            float sv1 = s[1][v] * scale;
            float sv2 = s[2][v] * scale;
            float sv3 = s[3][v] * scale;
            float mx = fmaxf(fmaxf(sv0, sv1), fmaxf(sv2, sv3));
#pragma unroll
            for (int off = 1; off < 16; off <<= 1)
                mx = fmaxf(mx, __shfl_xor(mx, off, 32));
            const float nm = fmaxf(rm[v], mx);
            const float corr = __expf(rm[v] - nm);
            float e0 = __expf(sv0 - nm);
            float e1 = __expf(sv1 - nm);
            float e2 = __expf(sv2 - nm);
            float e3 = __expf(sv3 - nm);
            float ps = (e0 + e1) + (e2 + e3);
#pragma unroll
            for (int off = 1; off < 16; off <<= 1)
                ps += __shfl_xor(ps, off, 32);
            rs[v] = rs[v] * corr + ps;
            rm[v] = nm;
            o[0][v] *= corr; o[1][v] *= corr; o[2][v] *= corr; o[3][v] *= corr;
            p[0][v] = e0; p[1][v] = e1; p[2][v] = e2; p[3][v] = e3;
        }

        // D-layout -> A-layout via per-wave LDS round trip (16x64 P tile)
#pragma unroll
        for (int v = 0; v < 8; ++v) {
            const int m = v + 8 * hl;
#pragma unroll
            for (int i = 0; i < 4; ++i)
                pw[m * 64 + i * 16 + nl] = (bf16)p[i][v];
        }
        const v16bf pa0 = load_frag_rm(pw, 64);        // keys kb..kb+31
        const v16bf pa1 = load_frag_rm(pw + 32, 64);   // keys kb+32..kb+63

        // O += P * V  (V^T tile row-major [64 dims, 64 keys])
#pragma unroll
        for (int dt = 0; dt < 4; ++dt) {
            v16bf vf = load_frag_rm(Vc + (size_t)(dt * 16) * 64, 64);
            o[dt] = WMMA_BF16(pa0, vf, o[dt]);
            vf = load_frag_rm(Vc + (size_t)(dt * 16) * 64 + 32, 64);
            o[dt] = WMMA_BF16(pa1, vf, o[dt]);
        }
    }

    // Normalize and store as bf16 [B, N, C]
    const int b = bh / 12;
    const int head = bh - b * 12;
#pragma unroll
    for (int v = 0; v < 8; ++v) {
        const float inv = 1.0f / rs[v];
        const size_t row = (size_t)b * 1024 + q0 + v + 8 * hl;
#pragma unroll
        for (int dt = 0; dt < 4; ++dt)
            attnb[row * 768 + head * 64 + dt * 16 + nl] = (bf16)(o[dt][v] * inv);
    }
}

// ---------------------------------------------------------------------------
// Proj GEMM: [16384, 768] x [768, 768]^T + bias -> f32 out.
// Same 32x64-per-wave blocking as the QKV GEMM.
// ---------------------------------------------------------------------------
__global__ __launch_bounds__(128) void proj_gemm_kernel(
        const bf16* __restrict__ attnb, const bf16* __restrict__ wb,
        const float* __restrict__ bias, float* __restrict__ out) {
    const int wave = threadIdx.x >> 5;
    const int lane = threadIdx.x & 31;
    const int m0 = blockIdx.x * 128 + wave * 32;
    const int n0 = blockIdx.y * 64;

    v8f acc[2][4] = {{v8f{}, v8f{}, v8f{}, v8f{}},
                     {v8f{}, v8f{}, v8f{}, v8f{}}};
    for (int k = 0; k < 768; k += 32) {
        v16bf a0 = load_frag_rm(attnb + (size_t)m0 * 768 + k, 768);
        v16bf a1 = load_frag_rm(attnb + (size_t)(m0 + 16) * 768 + k, 768);
#pragma unroll
        for (int nt = 0; nt < 4; ++nt) {
            v16bf b = load_frag_rm(wb + (size_t)(n0 + nt * 16) * 768 + k, 768);
            acc[0][nt] = WMMA_BF16(a0, b, acc[0][nt]);
            acc[1][nt] = WMMA_BF16(a1, b, acc[1][nt]);
        }
    }

    const int nl = lane & 15;
    const int hl = lane >> 4;
#pragma unroll
    for (int mt = 0; mt < 2; ++mt) {
#pragma unroll
        for (int nt = 0; nt < 4; ++nt) {
            const int c = n0 + nt * 16 + nl;
            const float bv = bias[c];
#pragma unroll
            for (int v = 0; v < 8; ++v) {
                const size_t r = (size_t)m0 + mt * 16 + v + 8 * hl;
                out[r * 768 + c] = acc[mt][nt][v] + bv;
            }
        }
    }
}

// ---------------------------------------------------------------------------
extern "C" void kernel_launch(void* const* d_in, const int* in_sizes, int n_in,
                              void* d_out, int out_size, void* d_ws, size_t ws_size,
                              hipStream_t stream) {
    const float* x      = (const float*)d_in[0];   // [16, 1024, 768]
    const float* w_qkv  = (const float*)d_in[1];   // [2304, 768]
    const float* w_proj = (const float*)d_in[2];   // [768, 768]
    const float* b_proj = (const float*)d_in[3];   // [768]
    float* out = (float*)d_out;                    // [16, 1024, 768]

    const size_t NX   = (size_t)16 * 1024 * 768;   // 12,582,912
    const size_t NWQ  = (size_t)2304 * 768;        //  1,769,472
    const size_t NWP  = (size_t)768 * 768;         //    589,824
    const size_t NQKV = (size_t)192 * 1024 * 64;   // 12,582,912 per tensor

    char* base = (char*)d_ws;
    size_t off = 0;
    auto carve = [&](size_t bytes) {
        char* p = base + off;
        off = (off + bytes + 255) & ~(size_t)255;
        return p;
    };
    bf16* xb     = (bf16*)carve(NX * 2);
    bf16* wqkvb  = (bf16*)carve(NWQ * 2);
    bf16* wprojb = (bf16*)carve(NWP * 2);
    bf16* Qb     = (bf16*)carve(NQKV * 2);
    bf16* Kb     = (bf16*)carve(NQKV * 2);
    bf16* Vtb    = (bf16*)carve(NQKV * 2);
    bf16* attnb  = (bf16*)carve(NX * 2);

    // 1) convert inputs to bf16
    cvt_bf16_kernel<<<(int)((NX  + 255) / 256), 256, 0, stream>>>(x, xb, (int)NX);
    cvt_bf16_kernel<<<(int)((NWQ + 255) / 256), 256, 0, stream>>>(w_qkv, wqkvb, (int)NWQ);
    cvt_bf16_kernel<<<(int)((NWP + 255) / 256), 256, 0, stream>>>(w_proj, wprojb, (int)NWP);

    // 2) QKV GEMM + head-major scatter (V transposed)
    qkv_gemm_kernel<<<dim3(128, 36), 128, 0, stream>>>(xb, wqkvb, Qb, Kb, Vtb);

    // 3) flash attention (async double-buffered K/V staging)
    attn_kernel<<<dim3(192, 16), 128, 0, stream>>>(Qb, Kb, Vtb, attnb);

    // 4) output projection + bias
    proj_gemm_kernel<<<dim3(128, 12), 128, 0, stream>>>(attnb, wprojb, b_proj, out);
}
